// Agent_Attention_82695300317573
// MI455X (gfx1250) — matrix operand order repaired and checked by reading, not compiled, not run on hardware
//
#include <hip/hip_runtime.h>
#include <hip/hip_bf16.h>

typedef __bf16 bf16;
typedef __attribute__((ext_vector_type(16))) __bf16 v16bf;
typedef __attribute__((ext_vector_type(8)))  float  v8f;

union FragAB {
    v16bf v;
    uint4 q[2];
    bf16  h[16];
};

__device__ __forceinline__ v8f wmma_bf16(v16bf a, v16bf b, v8f c) {
    // 8 args: (neg_a, A, neg_b, B, c_mod, C, reuse_a, reuse_b)
    return __builtin_amdgcn_wmma_f32_16x16x32_bf16(false, a, false, b, (short)0, c, false, false);
}

__device__ __forceinline__ void zero_frag(FragAB& f) {
    f.q[0] = make_uint4(0u, 0u, 0u, 0u);
    f.q[1] = make_uint4(0u, 0u, 0u, 0u);
}

// A fragment (16x32 bf16, MxK): lane-lo rows hold K {0..15}+k0, lane-hi {16..31}+k0.
__device__ __forceinline__ void load_a_row(FragAB& f, const bf16* rowp, int laneHi) {
    f.q[0] = *(const uint4*)(rowp + laneHi * 16);
    f.q[1] = *(const uint4*)(rowp + laneHi * 16 + 8);
}

// B fragment (32x16 bf16, KxN): lane n holds column n, K = laneHi*16 + e.
__device__ __forceinline__ void load_b_row(FragAB& f, const bf16* rowp, int laneHi) {
    const uint4* p = (const uint4*)(rowp + laneHi * 16);
    f.q[0] = p[0];
    f.q[1] = p[1];
}

// ---- CDNA5 async copy global->LDS (ASYNCcnt-tracked), 16 bytes per lane ----
__device__ __forceinline__ void async_ld_b128(void* lds_dst, const void* gsrc) {
    const unsigned lds_off = (unsigned)(uintptr_t)lds_dst;  // LDS aperture: addr[31:0] = LDS offset
    asm volatile("global_load_async_to_lds_b128 %0, %1, off" ::"v"(lds_off), "v"(gsrc)
                 : "memory");
}
__device__ __forceinline__ void wait_async0() {
    asm volatile("s_wait_asynccnt 0x0" ::: "memory");
}

// ---------------------------------------------------------------------------
// NT GEMM:  C[M,N] = A[M,K] * B[N,K]^T.   B is always bf16 (pre-converted).
// Block tile 256x128, K-step 32, 256 threads = 8 waves (4 M x 2 N),
// each wave owns a 64x64 tile = 4x4 WMMA accumulators (16 wmma / k-step).
// bf16 operands are streamed into LDS with global_load_async_to_lds_b128.
// ---------------------------------------------------------------------------
template <bool A_IS_F32, bool OUT_BF16, bool HAS_BIAS>
__global__ __launch_bounds__(256) void gemm_nt(const void* __restrict__ Ap, int lda,
                                               const bf16* __restrict__ Bp, int ldb,
                                               void* __restrict__ Cp, int ldc,
                                               const float* __restrict__ bias,
                                               int M, int N, int K) {
    __shared__ bf16 As[256 * 32];
    __shared__ bf16 Bs[128 * 32];

    const int tid    = threadIdx.x;
    const int lane   = tid & 31;
    const int wave   = tid >> 5;
    const int laneHi = lane >> 4;
    const int lane15 = lane & 15;
    const int waveM  = wave & 3;   // 0..3 -> 64-row slab
    const int waveN  = wave >> 2;  // 0..1 -> 64-col slab
    const int m0     = blockIdx.x * 256;
    const int n0     = blockIdx.y * 128;

    v8f acc[4][4];
#pragma unroll
    for (int i = 0; i < 4; ++i)
#pragma unroll
        for (int j = 0; j < 4; ++j) acc[i][j] = (v8f){0.f, 0.f, 0.f, 0.f, 0.f, 0.f, 0.f, 0.f};

    for (int k0 = 0; k0 < K; k0 += 32) {
        // ---- stage A tile: 256 rows x 32, one row per thread ----
        {
            const int gm = m0 + tid;
            bf16* dst    = &As[tid * 32];
            if (gm < M) {
                if constexpr (A_IS_F32) {
                    const float* src = (const float*)Ap + (size_t)gm * lda + k0;
                    __builtin_prefetch(src + 32, 0, 1);
                    float tmp[32];
                    const float4* s4 = (const float4*)src;
#pragma unroll
                    for (int i = 0; i < 8; ++i) ((float4*)tmp)[i] = s4[i];
#pragma unroll
                    for (int i = 0; i < 32; ++i) dst[i] = (bf16)tmp[i];
                } else {
                    const bf16* src = (const bf16*)Ap + (size_t)gm * lda + k0;
#pragma unroll
                    for (int i = 0; i < 4; ++i) async_ld_b128(dst + i * 8, src + i * 8);
                }
            } else {
                uint4* d4 = (uint4*)dst;
#pragma unroll
                for (int i = 0; i < 4; ++i) d4[i] = make_uint4(0u, 0u, 0u, 0u);
            }
        }
        // ---- stage B tile: 128 rows x 32, half row per thread (always bf16, async) ----
        {
            const int r  = tid >> 1;
            const int c  = (tid & 1) * 16;
            const int gn = n0 + r;
            bf16* dst    = &Bs[r * 32 + c];
            if (gn < N) {
                const bf16* src = Bp + (size_t)gn * ldb + k0 + c;
                async_ld_b128(dst, src);
                async_ld_b128(dst + 8, src + 8);
            } else {
                uint4* d4 = (uint4*)dst;
                d4[0] = d4[1] = make_uint4(0u, 0u, 0u, 0u);
            }
        }
        wait_async0();
        __syncthreads();

        FragAB a[4], b[4];
#pragma unroll
        for (int mt = 0; mt < 4; ++mt)
            load_a_row(a[mt], &As[(waveM * 64 + mt * 16 + lane15) * 32], laneHi);
#pragma unroll
        for (int nt = 0; nt < 4; ++nt)
            load_b_row(b[nt], &Bs[(waveN * 64 + nt * 16 + lane15) * 32], laneHi);
#pragma unroll
        for (int mt = 0; mt < 4; ++mt)
#pragma unroll
            for (int nt = 0; nt < 4; ++nt) acc[mt][nt] = wmma_bf16(a[mt].v, b[nt].v, acc[mt][nt]);
        __syncthreads();
    }

    // ---- epilogue: D element (m,n) lives in lane (m>=8)*16+n, vgpr m%8 ----
#pragma unroll
    for (int mt = 0; mt < 4; ++mt) {
#pragma unroll
        for (int nt = 0; nt < 4; ++nt) {
            const int mbase = m0 + waveM * 64 + mt * 16 + laneHi * 8;
            const int n     = n0 + waveN * 64 + nt * 16 + lane15;
            const float bv  = HAS_BIAS ? bias[n] : 0.0f;
#pragma unroll
            for (int r = 0; r < 8; ++r) {
                const int m = mbase + r;
                if (m < M) {
                    const float val = acc[mt][nt][r] + bv;
                    if constexpr (OUT_BF16)
                        ((bf16*)Cp)[(size_t)m * ldc + n] = (bf16)val;
                    else
                        ((float*)Cp)[(size_t)m * ldc + n] = val;
                }
            }
        }
    }
}

// ---------------------------------------------------------------------------
__global__ __launch_bounds__(256) void cvt_f32_to_bf16(const float* __restrict__ s,
                                                       bf16* __restrict__ d, int n) {
    for (int i = blockIdx.x * 256 + threadIdx.x; i < n; i += gridDim.x * 256) d[i] = (bf16)s[i];
}

// ---------------------------------------------------------------------------
// Adaptive 24x24 -> 7x7 average pool of q spatial tokens -> router [B,49,C] bf16
// ---------------------------------------------------------------------------
__global__ __launch_bounds__(256) void pool_router(const bf16* __restrict__ qkv,
                                                   bf16* __restrict__ router) {
    const int b  = blockIdx.x;
    const int a  = blockIdx.y;  // 0..48
    const int ay = a / 7, ax = a % 7;
    const int ys = (ay * 24) / 7, ye = ((ay + 1) * 24 + 6) / 7;
    const int xs = (ax * 24) / 7, xe = ((ax + 1) * 24 + 6) / 7;
    const float inv = 1.0f / (float)((ye - ys) * (xe - xs));
    for (int c = threadIdx.x; c < 768; c += 256) {
        float s = 0.0f;
        for (int y = ys; y < ye; ++y)
            for (int x = xs; x < xe; ++x) {
                const int t = y * 24 + x;  // spatial tokens 0..575 (class token is last)
                s += (float)qkv[((size_t)b * 577 + t) * 2304 + c];
            }
        router[((size_t)b * 49 + a) * 768 + c] = (bf16)(s * inv);
    }
}

// ---------------------------------------------------------------------------
// Stage 1a: s1[bh, 49, 592] = (router_h [49x64]) x (k_h^T [64x577]) * scale
// ---------------------------------------------------------------------------
__global__ __launch_bounds__(256) void stage1_scores(const bf16* __restrict__ qkv,
                                                     const bf16* __restrict__ router,
                                                     float* __restrict__ s1ws) {
    const int bh = blockIdx.x;
    const int b = bh / 12, h = bh % 12;
    const int lane = threadIdx.x & 31, wave = threadIdx.x >> 5;
    const int laneHi = lane >> 4, lane15 = lane & 15;
    const bf16* rbase = router + ((size_t)b * 49) * 768 + h * 64;
    const bf16* kbase = qkv + ((size_t)b * 577) * 2304 + 768 + h * 64;
    float* out = s1ws + (size_t)bh * 49 * 592;

    for (int tile = wave; tile < 4 * 37; tile += 8) {
        const int mt = tile & 3, nt = tile >> 2;
        v8f acc = (v8f){0.f, 0.f, 0.f, 0.f, 0.f, 0.f, 0.f, 0.f};
#pragma unroll
        for (int k0 = 0; k0 < 64; k0 += 32) {
            FragAB a, bfr;
            const int arow = mt * 16 + lane15;
            if (arow < 49)
                load_a_row(a, rbase + (size_t)arow * 768 + k0, laneHi);
            else
                zero_frag(a);
            const int tok = nt * 16 + lane15;
            if (tok < 577)
                load_b_row(bfr, kbase + (size_t)tok * 2304 + k0, laneHi);
            else
                zero_frag(bfr);
            acc = wmma_bf16(a.v, bfr.v, acc);
        }
        const int n = nt * 16 + lane15;
#pragma unroll
        for (int r = 0; r < 8; ++r) {
            const int m = mt * 16 + laneHi * 8 + r;
            if (m < 49) out[(size_t)m * 592 + n] = acc[r] * 0.125f;  // 64^-0.5
        }
    }
}

// ---------------------------------------------------------------------------
// Stage 1b: row softmax over 577 tokens -> probs bf16 [bh, 49, 608] (zero-padded K)
// ---------------------------------------------------------------------------
__global__ __launch_bounds__(256) void stage1_softmax(const float* __restrict__ s1ws,
                                                      bf16* __restrict__ probs) {
    const int bh = blockIdx.x;
    const float* S = s1ws + (size_t)bh * 49 * 592;
    bf16* P = probs + (size_t)bh * 49 * 608;
    const int lane = threadIdx.x & 31, wave = threadIdx.x >> 5;
    for (int r = wave; r < 49; r += 8) {
        const float* row = S + (size_t)r * 592;
        float mx = -1e30f;
        for (int c = lane; c < 577; c += 32) mx = fmaxf(mx, row[c]);
#pragma unroll
        for (int o = 16; o > 0; o >>= 1) mx = fmaxf(mx, __shfl_xor(mx, o, 32));
        float sum = 0.0f;
        for (int c = lane; c < 577; c += 32) sum += __expf(row[c] - mx);
#pragma unroll
        for (int o = 16; o > 0; o >>= 1) sum += __shfl_xor(sum, o, 32);
        const float rinv = 1.0f / sum;
        bf16* prow = P + (size_t)r * 608;
        for (int c = lane; c < 608; c += 32)
            prow[c] = (c < 577) ? (bf16)(__expf(row[c] - mx) * rinv) : (bf16)0.0f;
    }
}

// ---------------------------------------------------------------------------
// Stage 1c: rv[b,49,C] (bf16) = probs [49x608] x v_h [608(->577)x64]
// ---------------------------------------------------------------------------
__global__ __launch_bounds__(256) void stage1_rv(const bf16* __restrict__ qkv,
                                                 const bf16* __restrict__ probs,
                                                 bf16* __restrict__ rv) {
    const int bh = blockIdx.x;
    const int b = bh / 12, h = bh % 12;
    const int lane = threadIdx.x & 31, wave = threadIdx.x >> 5;
    const int laneHi = lane >> 4, lane15 = lane & 15;
    const bf16* P     = probs + (size_t)bh * 49 * 608;
    const bf16* vbase = qkv + ((size_t)b * 577) * 2304 + 1536 + h * 64;
    bf16* out         = rv + ((size_t)b * 49) * 768 + h * 64;

    for (int tile = wave; tile < 16; tile += 8) {
        const int mt = tile & 3, nt = tile >> 2;
        v8f acc = (v8f){0.f, 0.f, 0.f, 0.f, 0.f, 0.f, 0.f, 0.f};
        const int d = nt * 16 + lane15;
        for (int k0 = 0; k0 < 608; k0 += 32) {
            FragAB a, bfr;
            const int arow = mt * 16 + lane15;
            if (arow < 49)
                load_a_row(a, P + (size_t)arow * 608 + k0, laneHi);
            else
                zero_frag(a);
#pragma unroll
            for (int e = 0; e < 16; ++e) {  // strided column of v_h
                const int tok = k0 + laneHi * 16 + e;
                bfr.h[e] = (tok < 577) ? vbase[(size_t)tok * 2304 + d] : (bf16)0.0f;
            }
            acc = wmma_bf16(a.v, bfr.v, acc);
        }
#pragma unroll
        for (int r = 0; r < 8; ++r) {
            const int m = mt * 16 + laneHi * 8 + r;
            if (m < 49) out[(size_t)m * 768 + d] = (bf16)acc[r];
        }
    }
}

// ---------------------------------------------------------------------------
// Stage 2 (fused): per (b,h, 64-token chunk):
//   s2 = q_h[64x64] x router_h^T [64x49] -> softmax(49) -> x rv_h [49x64]
// ---------------------------------------------------------------------------
__global__ __launch_bounds__(256) void stage2(const bf16* __restrict__ qkv,
                                              const bf16* __restrict__ router,
                                              const bf16* __restrict__ rv,
                                              bf16* __restrict__ attn) {
    __shared__ float s2[64 * 64];
    __shared__ bf16 pr[64 * 64];

    const int id = blockIdx.x;
    const int rc = id % 10;
    const int bh = id / 10;
    const int b = bh / 12, h = bh % 12;
    const int row0 = rc * 64;
    const int tid = threadIdx.x;
    const int lane = tid & 31, wave = tid >> 5;
    const int laneHi = lane >> 4, lane15 = lane & 15;

    const bf16* qbase  = qkv + ((size_t)b * 577) * 2304 + h * 64;
    const bf16* rbase  = router + ((size_t)b * 49) * 768 + h * 64;
    const bf16* rvbase = rv + ((size_t)b * 49) * 768 + h * 64;
    bf16* obase        = attn + ((size_t)b * 577) * 768 + h * 64;

    for (int tile = wave; tile < 16; tile += 8) {
        const int mt = tile & 3, nt = tile >> 2;
        v8f acc = (v8f){0.f, 0.f, 0.f, 0.f, 0.f, 0.f, 0.f, 0.f};
#pragma unroll
        for (int k0 = 0; k0 < 64; k0 += 32) {
            FragAB a, bfr;
            const int tok = row0 + mt * 16 + lane15;
            if (tok < 577)
                load_a_row(a, qbase + (size_t)tok * 2304 + k0, laneHi);
            else
                zero_frag(a);
            const int ag = nt * 16 + lane15;
            if (ag < 49)
                load_b_row(bfr, rbase + (size_t)ag * 768 + k0, laneHi);
            else
                zero_frag(bfr);
            acc = wmma_bf16(a.v, bfr.v, acc);
        }
        const int n = nt * 16 + lane15;
#pragma unroll
        for (int r = 0; r < 8; ++r) s2[(mt * 16 + laneHi * 8 + r) * 64 + n] = acc[r] * 0.125f;
    }
    __syncthreads();

    if (tid < 64) {
        float mx = -1e30f;
        for (int c = 0; c < 49; ++c) mx = fmaxf(mx, s2[tid * 64 + c]);
        float sum = 0.0f;
        for (int c = 0; c < 49; ++c) sum += __expf(s2[tid * 64 + c] - mx);
        const float rinv = 1.0f / sum;
        for (int c = 0; c < 64; ++c)
            pr[tid * 64 + c] = (c < 49) ? (bf16)(__expf(s2[tid * 64 + c] - mx) * rinv) : (bf16)0.0f;
    }
    __syncthreads();

    for (int tile = wave; tile < 16; tile += 8) {
        const int mt = tile & 3, nt = tile >> 2;
        v8f acc = (v8f){0.f, 0.f, 0.f, 0.f, 0.f, 0.f, 0.f, 0.f};
        const int d = nt * 16 + lane15;
#pragma unroll
        for (int k0 = 0; k0 < 64; k0 += 32) {
            FragAB a, bfr;
            load_a_row(a, &pr[(mt * 16 + lane15) * 64 + k0], laneHi);
#pragma unroll
            for (int e = 0; e < 16; ++e) {  // strided column of rv_h
                const int ag = k0 + laneHi * 16 + e;
                bfr.h[e] = (ag < 49) ? rvbase[(size_t)ag * 768 + d] : (bf16)0.0f;
            }
            acc = wmma_bf16(a.v, bfr.v, acc);
        }
#pragma unroll
        for (int r = 0; r < 8; ++r) {
            const int tok = row0 + mt * 16 + laneHi * 8 + r;
            if (tok < 577) obase[(size_t)tok * 768 + d] = (bf16)acc[r];
        }
    }
}

// ---------------------------------------------------------------------------
extern "C" void kernel_launch(void* const* d_in, const int* in_sizes, int n_in,
                              void* d_out, int out_size, void* d_ws, size_t ws_size,
                              hipStream_t stream) {
    (void)in_sizes; (void)n_in; (void)out_size; (void)ws_size;
    const float* x     = (const float*)d_in[0];  // [64,577,768]
    const float* Wqkv  = (const float*)d_in[1];  // [2304,768]
    const float* Wproj = (const float*)d_in[2];  // [768,768]
    const float* bproj = (const float*)d_in[3];  // [768]
    float* out = (float*)d_out;                  // [64,577,768]

    const int M = 64 * 577;  // 36928 rows
    char* ws = (char*)d_ws;
    size_t off = 0;
    auto alloc = [&](size_t bytes) -> void* {
        void* p = ws + off;
        off += (bytes + 255) & ~(size_t)255;
        return p;
    };
    bf16* qkvb    = (bf16*)alloc((size_t)M * 2304 * 2);        // [M,2304] bf16 (q|k|v)
    bf16* wqkvb   = (bf16*)alloc((size_t)2304 * 768 * 2);
    bf16* wprojb  = (bf16*)alloc((size_t)768 * 768 * 2);
    bf16* routerb = (bf16*)alloc((size_t)64 * 49 * 768 * 2);   // [B,49,C]
    bf16* rvb     = (bf16*)alloc((size_t)64 * 49 * 768 * 2);   // [B,49,C]
    bf16* attnb   = (bf16*)alloc((size_t)M * 768 * 2);         // [M,C]
    float* s1ws   = (float*)alloc((size_t)768 * 49 * 592 * 4); // per-(b,h) scores
    bf16* probsws = (bf16*)alloc((size_t)768 * 49 * 608 * 2);  // per-(b,h) probs

    dim3 blk(256);
    cvt_f32_to_bf16<<<dim3((2304 * 768 + 255) / 256), blk, 0, stream>>>(Wqkv, wqkvb, 2304 * 768);
    cvt_f32_to_bf16<<<dim3((768 * 768 + 255) / 256), blk, 0, stream>>>(Wproj, wprojb, 768 * 768);
    // qkv = x @ Wqkv^T  (f32 A converted in LDS fill, bf16 B via async-to-LDS)
    gemm_nt<true, true, false>
        <<<dim3((M + 255) / 256, 2304 / 128), blk, 0, stream>>>(x, 768, wqkvb, 768, qkvb, 2304,
                                                                nullptr, M, 2304, 768);
    pool_router<<<dim3(64, 49), blk, 0, stream>>>(qkvb, routerb);
    stage1_scores<<<768, blk, 0, stream>>>(qkvb, routerb, s1ws);
    stage1_softmax<<<768, blk, 0, stream>>>(s1ws, probsws);
    stage1_rv<<<768, blk, 0, stream>>>(qkvb, probsws, rvb);
    stage2<<<7680, blk, 0, stream>>>(qkvb, routerb, rvb, attnb);
    // out = attn @ Wproj^T + b  (bf16 A and B via async-to-LDS, f32 out + bias)
    gemm_nt<false, false, true>
        <<<dim3((M + 255) / 256, 768 / 128), blk, 0, stream>>>(attnb, 768, wprojb, 768, out, 768,
                                                               bproj, M, 768, 768);
}
